// HIN_MLP_31258771980727
// MI455X (gfx1250) — compile-verified, matrix operand order checked by reading, not run on hardware
//
#include <hip/hip_runtime.h>

// ---------------------------------------------------------------------------
// Fused gather + 2-layer MLP for MI455X (gfx1250), wave32, bf16 WMMA.
//   x   = emb[idx0, idx1, :]                (B x 128)
//   h   = relu(x @ w1^T + b1)               (B x 256)   -- kept in LDS
//   out = relu(h @ w2^T + b2)               (B x 86)
// One workgroup = 256 threads = 8 waves, processes 64 batch rows.
// ---------------------------------------------------------------------------

typedef __attribute__((ext_vector_type(16))) __bf16 v16bf;
typedef __attribute__((ext_vector_type(8)))  __bf16 v8bf;
typedef __attribute__((ext_vector_type(8)))  float  v8f;

#define FEAT   128
#define NHID   256
#define NREL   86
#define NRELP  96        // 86 padded to 6 tiles of 16
#define ROWS_WG 64       // batch rows per workgroup

// padded LDS strides (in bf16 elements) to avoid 64-bank conflicts:
// stride*2 bytes == 16B-aligned and shifts banks by 4 per row.
#define W1S 136          // w1 LDS: [NHID][W1S]
#define W2S 264          // w2 LDS: [NRELP][W2S]
#define HS  264          // h  LDS: [ROWS_WG][HS]

// pack two float4 (8 consecutive K values) into half of an A/B fragment
template <int BASE>
__device__ __forceinline__ void pack8(v16bf& a, float4 u, float4 v) {
  a[BASE + 0] = (__bf16)u.x; a[BASE + 1] = (__bf16)u.y;
  a[BASE + 2] = (__bf16)u.z; a[BASE + 3] = (__bf16)u.w;
  a[BASE + 4] = (__bf16)v.x; a[BASE + 5] = (__bf16)v.y;
  a[BASE + 6] = (__bf16)v.z; a[BASE + 7] = (__bf16)v.w;
}

// Load a 16-lane-row x 32-K bf16 fragment slice for this lane from LDS.
// base already points at row*stride + hi*8; the two 8-half chunks sit at
// +0 and +16 K (ISA 7.12.2 16-bit A/B layout). Both are 16B aligned.
__device__ __forceinline__ v16bf ldsFrag(const __bf16* base) {
  v8bf lo = *(const v8bf*)(base);
  v8bf hi = *(const v8bf*)(base + 16);
  v16bf r;
#pragma unroll
  for (int i = 0; i < 8; ++i) { r[i] = lo[i]; r[i + 8] = hi[i]; }
  return r;
}

__global__ __launch_bounds__(256) void hin_mlp_fused(
    const float* __restrict__ emb,   // [500000][4][128]
    const float* __restrict__ w1,    // [256][128]  (N x K)
    const float* __restrict__ b1,    // [256]
    const float* __restrict__ w2,    // [86][256]   (N x K)
    const float* __restrict__ b2,    // [86]
    const int*   __restrict__ idx0,  // [B]
    const int*   __restrict__ idx1,  // [B]
    float*       __restrict__ out)   // [B][86]
{
  __shared__ __bf16 w1_lds[NHID * W1S];
  __shared__ __bf16 w2_lds[NRELP * W2S];
  __shared__ __bf16 h_lds[ROWS_WG * HS];
  __shared__ float  b1_lds[NHID];
  __shared__ float  b2_lds[NRELP];

  const int tid   = threadIdx.x;
  const int wave  = tid >> 5;
  const int lane  = tid & 31;
  const int row16 = lane & 15;   // row within a 16-row tile
  const int hi    = lane >> 4;   // which K half-group this lane holds

  // ---- stage weights/biases f32 -> bf16 into LDS -------------------------
  for (int i = tid; i < NHID * FEAT; i += 256) {
    int n = i >> 7, k = i & 127;
    w1_lds[n * W1S + k] = (__bf16)w1[i];
  }
  for (int i = tid; i < NRELP * NHID; i += 256) {
    int n = i >> 8, k = i & 255;
    float v = (n < NREL) ? w2[n * NHID + k] : 0.0f;
    w2_lds[n * W2S + k] = (__bf16)v;
  }
  if (tid < NHID)  b1_lds[tid] = b1[tid];
  if (tid < NRELP) b2_lds[tid] = (tid < NREL) ? b2[tid] : 0.0f;

  // ---- layer 1: gather A fragments straight from global ------------------
  const int mtile = wave & 3;  // 4 M-tiles of 16 rows, 2 waves share each
  const int brow  = blockIdx.x * ROWS_WG + mtile * 16 + row16;
  const size_t src =
      ((size_t)(unsigned)idx0[brow] * 4u + (unsigned)idx1[brow]) * (size_t)FEAT;

  v16bf a1[4];
#pragma unroll
  for (int s = 0; s < 4; ++s) {
    const float4* p = (const float4*)(emb + src + s * 32 + hi * 8);
    pack8<0>(a1[s], p[0], p[1]);   // K = s*32 + hi*8 .. +7
    pack8<8>(a1[s], p[4], p[5]);   // K = s*32 + hi*8 + 16 .. +23
  }

  __syncthreads();  // weights staged

  // waves 0-3: N tiles 0..7 ; waves 4-7: N tiles 8..15
  const int nt0 = (wave >> 2) * 8;
  for (int nt = nt0; nt < nt0 + 8; ++nt) {
    v8f c = {};
    const __bf16* wrow = &w1_lds[(nt * 16 + row16) * W1S + hi * 8];
#pragma unroll
    for (int s = 0; s < 4; ++s) {
      v16bf bf = ldsFrag(wrow + s * 32);
      c = __builtin_amdgcn_wmma_f32_16x16x32_bf16(
              false, a1[s], false, bf, (short)0, c, false, false);
    }
    // C layout: lane 0-15 -> M=j, lane 16-31 -> M=j+8 ; N = row16
    const int   col  = nt * 16 + row16;
    const float bias = b1_lds[col];
    const int   hr0  = mtile * 16 + hi * 8;
#pragma unroll
    for (int j = 0; j < 8; ++j) {
      float v = c[j] + bias;
      h_lds[(hr0 + j) * HS + col] = (__bf16)(v > 0.0f ? v : 0.0f);
    }
  }

  __syncthreads();  // h fully materialized in LDS

  // ---- layer 2: h (LDS, bf16) @ w2^T ------------------------------------
  v16bf a2[8];
  const __bf16* hrow = &h_lds[(mtile * 16 + row16) * HS + hi * 8];
#pragma unroll
  for (int s = 0; s < 8; ++s) a2[s] = ldsFrag(hrow + s * 32);

  // waves 0-3: N tiles 0..2 ; waves 4-7: N tiles 3..5  (6 tiles = 96 cols)
  const int nt2 = (wave >> 2) * 3;
  for (int nt = nt2; nt < nt2 + 3; ++nt) {
    v8f c = {};
    const __bf16* wrow2 = &w2_lds[(nt * 16 + row16) * W2S + hi * 8];
#pragma unroll
    for (int s = 0; s < 8; ++s) {
      v16bf bf = ldsFrag(wrow2 + s * 32);
      c = __builtin_amdgcn_wmma_f32_16x16x32_bf16(
              false, a2[s], false, bf, (short)0, c, false, false);
    }
    const int col = nt * 16 + row16;
    if (col < NREL) {
      const float  bias = b2_lds[col];
      const size_t ob   =
          (size_t)(blockIdx.x * ROWS_WG + mtile * 16 + hi * 8) * NREL + col;
#pragma unroll
      for (int j = 0; j < 8; ++j) {
        float v = c[j] + bias;
        out[ob + (size_t)j * NREL] = v > 0.0f ? v : 0.0f;
      }
    }
  }
}

extern "C" void kernel_launch(void* const* d_in, const int* in_sizes, int n_in,
                              void* d_out, int out_size, void* d_ws, size_t ws_size,
                              hipStream_t stream) {
  const float* emb  = (const float*)d_in[0];
  const float* w1   = (const float*)d_in[1];
  const float* b1   = (const float*)d_in[2];
  const float* w2   = (const float*)d_in[3];
  const float* b2   = (const float*)d_in[4];
  const int*   idx0 = (const int*)d_in[5];
  const int*   idx1 = (const int*)d_in[6];
  float*       out  = (float*)d_out;

  const int B    = in_sizes[5];        // 131072
  const int grid = B / ROWS_WG;        // 2048 workgroups

  hipLaunchKernelGGL(hin_mlp_fused, dim3(grid), dim3(256), 0, stream,
                     emb, w1, b1, w2, b2, idx0, idx1, out);
}
